// LSTM_50208167690847
// MI455X (gfx1250) — matrix operand order repaired and checked by reading, not compiled, not run on hardware
//
#include <hip/hip_runtime.h>
#include <hip/hip_bf16.h>

// ---------------------------------------------------------------------------
// Bidirectional 2-layer LSTM (B=256, T=512, IN=64, H=256, G=4H=1024) + FC.
// Batch-parallel recurrence (16-row WMMA M-tiles per WG, no grid sync), fused
// [Whh|Wih] x [h_t|x_t] bf16 WMMA per step, fp32 gates, single-step layer-1
// backward. Weight base pointer is laundered through empty inline asm each
// step so LICM cannot hoist 256 VGPRs of loop-invariant weight tiles out of
// the T-loop (which caused scratch spill/reload on the critical path).
// ---------------------------------------------------------------------------

typedef __attribute__((ext_vector_type(16))) __bf16 v16bf;
typedef __attribute__((ext_vector_type(8)))  float  v8f;

constexpr int Bb = 256;   // batch
constexpr int Tt = 512;   // time
constexpr int INx = 64;   // layer0 input features
constexpr int Hh = 256;   // hidden
constexpr int Gg = 1024;  // 4*H gate columns
constexpr int NTILES = Gg / 16;  // 64
constexpr int KTH = Hh / 32;               // 8 K-tiles for the h part
constexpr int KT0 = (Hh + INx) / 32;       // 10 K-tiles, layer 0
constexpr int KT1 = (Hh + 2 * Hh) / 32;    // 24 K-tiles, layer 1
constexpr int XKT0 = INx / 32;             // 2
constexpr int XKT1 = (2 * Hh) / 32;        // 16

// ---- WMMA bf16 register-layout index helpers (CDNA5 ISA 7.12.2) --
__device__ __forceinline__ int a_k_of(int lane, int idx) {
  int j = idx >> 1, h = idx & 1;
  return ((lane >> 4) << 3) | ((j >> 2) << 4) | ((j & 3) << 1) | h;
}
__device__ __forceinline__ int a_slot_of(int m, int k) {
  int grp = k >> 4;
  int rem = k & 15;
  int lh  = rem >> 3;
  int w   = rem & 7;
  int jl  = w >> 1;
  int h   = w & 1;
  return ((lh << 4) | m) * 16 + (((grp << 2) | jl) << 1) + h;
}
__device__ __forceinline__ int b_k_of(int lane, int idx) {
  int j = idx >> 1, h = idx & 1;
  return ((lane >> 4) << 3) | ((j >> 2) << 4) | ((j & 3) << 1) | h;
}

__device__ __forceinline__ float sigf(float x) {
  return 1.0f / (1.0f + __expf(-x));
}
__device__ __forceinline__ float tanh_fast(float x) {
  float e = __expf(-2.0f * x);
  return (1.0f - e) / (1.0f + e);
}

// Make a pointer opaque to LICM/alias analysis (no code emitted).
__device__ __forceinline__ const __bf16* launder_ptr(const __bf16* p) {
  unsigned long long v = (unsigned long long)p;
  asm volatile("" : "+v"(v));
  return (const __bf16*)v;
}

// ---------------------------------------------------------------------------
// Pack [Whh | Wih] (fp32) -> bf16 B-operand tiles: dst[dir][ntile][ktile][512]
// ---------------------------------------------------------------------------
__global__ void pack_w_kernel(const float* __restrict__ Whh,
                              const float* __restrict__ Wih,
                              __bf16* __restrict__ dst,
                              int Kt, int IN, int total) {
  for (int i = blockIdx.x * blockDim.x + threadIdx.x; i < total;
       i += gridDim.x * blockDim.x) {
    int v = i & 511;
    int lane = v >> 4, idx = v & 15;
    int rest = i >> 9;
    int kt = rest % Kt; rest /= Kt;
    int nt = rest & 63;
    int dir = rest >> 6;
    int g = nt * 16 + (lane & 15);
    int k = kt * 32 + b_k_of(lane, idx);
    float val;
    if (k < Hh) val = Whh[((size_t)dir * Gg + g) * Hh + k];
    else        val = Wih[((size_t)dir * Gg + g) * IN + (k - Hh)];
    dst[i] = (__bf16)val;
  }
}

// ---------------------------------------------------------------------------
// Pack x (fp32 [B,T,IN]) -> bf16 A-operand tiles: dst[t][btile][xkt][512]
// ---------------------------------------------------------------------------
__global__ void pack_x0_kernel(const float* __restrict__ x,
                               __bf16* __restrict__ dst, int total) {
  for (int i = blockIdx.x * blockDim.x + threadIdx.x; i < total;
       i += gridDim.x * blockDim.x) {
    int v = i & 511;
    int lane = v >> 4, idx = v & 15;
    int xkt = (i >> 9) & (XKT0 - 1);
    int bt  = (i >> 10) & 15;
    int t   = i >> 14;
    int m = lane & 15;
    int k = xkt * 32 + a_k_of(lane, idx);       // 0..63
    int b = bt * 16 + m;
    dst[i] = (__bf16)x[((size_t)b * Tt + t) * INx + k];
  }
}

// ---------------------------------------------------------------------------
// One bidirectional LSTM layer, fully fused recurrence.
// grid = (16 batch tiles, 2 directions), block = 512 threads (16 waves).
// ---------------------------------------------------------------------------
__global__ void __launch_bounds__(512)
lstm_layer_kernel(const __bf16* __restrict__ Wpack,   // [2][64][Kt][512]
                  const __bf16* __restrict__ Xpack,   // [T][16][XKT][512]
                  const float*  __restrict__ bias,    // [2][1024]
                  __bf16* __restrict__ OutPack,       // [T][16][16][512] or null
                  float*  __restrict__ OutLast,       // [2][256][256] or null
                  int Kt, int XKT, int steps_fwd, int steps_bwd) {
  const int tid  = threadIdx.x;
  const int lane = tid & 31;
  const int wv   = tid >> 5;              // wave 0..15
  const int btile = blockIdx.x;           // 0..15
  const int dir   = blockIdx.y;           // 0 fwd, 1 bwd
  const int steps = (dir == 0) ? steps_fwd : steps_bwd;
  const int xkt_n = Kt - KTH;             // #x K-tiles (2 or 16)

  __shared__ __bf16 apack[KTH * 512];     // packed h A-tiles, 8KB
  __shared__ float  gbuf[16 * 1024];      // gate pre-activations, 64KB

  for (int i = tid; i < KTH * 512; i += 512) apack[i] = (__bf16)0.0f;
  __syncthreads();

  // Pointwise ownership: thread handles elements e = tid*8 .. tid*8+7
  const int e0 = tid * 8;
  const int m  = e0 >> 8;
  const int j0 = e0 & 255;
  float creg[8];
  float bi[8], bf2[8], bg[8], bo[8];
  const float* bp = bias + (size_t)dir * Gg;
  #pragma unroll
  for (int q = 0; q < 8; ++q) {
    creg[q] = 0.0f;
    int j = j0 + q;
    bi[q]  = bp[j];
    bf2[q] = bp[256 + j];
    bg[q]  = bp[512 + j];
    bo[q]  = bp[768 + j];
  }

  // This wave's 4 consecutive N-tiles of packed weights, h-part first.
  const __bf16* wb0 =
      Wpack + ((size_t)dir * NTILES + (size_t)wv * 4) * Kt * 512 + lane * 16;
  const size_t wstride = (size_t)Kt * 512;
  const __bf16* apl = apack + lane * 16;

  for (int step = 0; step < steps; ++step) {
    const int t = (dir == 0) ? step : (Tt - 1 - step);
    const __bf16* xb =
        Xpack + (((size_t)t * 16 + btile) * XKT) * 512 + lane * 16;

    // Prefetch next step's x tiles while this step computes.
    if (step + 1 < steps) {
      const int tn = (dir == 0) ? (t + 1) : (t - 1);
      const __bf16* xn =
          Xpack + (((size_t)tn * 16 + btile) * XKT) * 512 + lane * 16;
      __builtin_prefetch(xn, 0, 3);
    }

    // Opaque per-step weight pointer: keeps weight loads inside the T-loop
    // (streamed from L2) instead of being hoisted + spilled to scratch.
    const __bf16* wls = launder_ptr(wb0);

    v8f acc0 = {}, acc1 = {}, acc2 = {}, acc3 = {};

    // Preload all 8 recurrent A-tiles (h) from LDS in one burst.
    v16bf ah[KTH];
    #pragma unroll
    for (int kt = 0; kt < KTH; ++kt)
      ah[kt] = *(const v16bf*)(apl + kt * 512);

    // --- recurrent part ---
    #pragma unroll
    for (int kt = 0; kt < KTH; ++kt) {
      const __bf16* wb = wls + (size_t)kt * 512;
      v16bf b0v = *(const v16bf*)(wb);
      v16bf b1v = *(const v16bf*)(wb + wstride);
      v16bf b2v = *(const v16bf*)(wb + 2 * wstride);
      v16bf b3v = *(const v16bf*)(wb + 3 * wstride);
      acc0 = __builtin_amdgcn_wmma_f32_16x16x32_bf16(false, ah[kt], false, b0v,
                                                     (short)0, acc0, false, false);
      acc1 = __builtin_amdgcn_wmma_f32_16x16x32_bf16(false, ah[kt], false, b1v,
                                                     (short)0, acc1, false, false);
      acc2 = __builtin_amdgcn_wmma_f32_16x16x32_bf16(false, ah[kt], false, b2v,
                                                     (short)0, acc2, false, false);
      acc3 = __builtin_amdgcn_wmma_f32_16x16x32_bf16(false, ah[kt], false, b3v,
                                                     (short)0, acc3, false, false);
    }

    // --- input part: A from global x-pack ---
    for (int xk = 0; xk < xkt_n; ++xk) {
      v16bf a = *(const v16bf*)(xb + (size_t)xk * 512);
      const __bf16* wb = wls + (size_t)(KTH + xk) * 512;
      v16bf b0v = *(const v16bf*)(wb);
      v16bf b1v = *(const v16bf*)(wb + wstride);
      v16bf b2v = *(const v16bf*)(wb + 2 * wstride);
      v16bf b3v = *(const v16bf*)(wb + 3 * wstride);
      acc0 = __builtin_amdgcn_wmma_f32_16x16x32_bf16(false, a, false, b0v,
                                                     (short)0, acc0, false, false);
      acc1 = __builtin_amdgcn_wmma_f32_16x16x32_bf16(false, a, false, b1v,
                                                     (short)0, acc1, false, false);
      acc2 = __builtin_amdgcn_wmma_f32_16x16x32_bf16(false, a, false, b2v,
                                                     (short)0, acc2, false, false);
      acc3 = __builtin_amdgcn_wmma_f32_16x16x32_bf16(false, a, false, b3v,
                                                     (short)0, acc3, false, false);
    }

    // Scatter accumulators to LDS gate buffer.
    {
      const int mb = (lane >> 4) * 8;
      const int nc = lane & 15;
      #pragma unroll
      for (int r = 0; r < 8; ++r) {
        int row = mb + r;
        gbuf[row * 1024 + (wv * 4 + 0) * 16 + nc] = acc0[r];
        gbuf[row * 1024 + (wv * 4 + 1) * 16 + nc] = acc1[r];
        gbuf[row * 1024 + (wv * 4 + 2) * 16 + nc] = acc2[r];
        gbuf[row * 1024 + (wv * 4 + 3) * 16 + nc] = acc3[r];
      }
    }
    __syncthreads();

    // LSTM pointwise: i,f,g,o -> c,h ; repack h (bf16) into A layout.
    {
      const float* gr = gbuf + m * 1024;
      #pragma unroll
      for (int q = 0; q < 8; ++q) {
        int j = j0 + q;
        float ig = sigf(gr[j] + bi[q]);
        float fg = sigf(gr[256 + j] + bf2[q]);
        float gc = tanh_fast(gr[512 + j] + bg[q]);
        float og = sigf(gr[768 + j] + bo[q]);
        float c  = fg * creg[q] + ig * gc;
        creg[q]  = c;
        float h  = og * tanh_fast(c);
        __bf16 hb = (__bf16)h;
        apack[(j >> 5) * 512 + a_slot_of(m, j & 31)] = hb;
        if (OutPack) {
          int col = dir * Hh + j;     // fwd -> cols 0..255, bwd -> 256..511
          OutPack[(((size_t)t * 16 + btile) * 16 + (col >> 5)) * 512 +
                  a_slot_of(m, col & 31)] = hb;
        }
        if (OutLast && step == steps - 1) {
          OutLast[((size_t)dir * Bb + btile * 16 + m) * Hh + j] = h;
        }
      }
    }
    __syncthreads();
  }
}

// ---------------------------------------------------------------------------
// Final FC: out[b] = dot(concat(h1f[b], h1b[b]), fc_w) + fc_b
// ---------------------------------------------------------------------------
__global__ void fc_kernel(const float* __restrict__ h1,   // [2][256][256]
                          const float* __restrict__ fcw,  // [512]
                          const float* __restrict__ fcb,  // [1]
                          float* __restrict__ out) {      // [256]
  int b = threadIdx.x;
  float s = fcb[0];
  const float* hf  = h1 + (size_t)b * Hh;
  const float* hbk = h1 + (size_t)Bb * Hh + (size_t)b * Hh;
  for (int j = 0; j < Hh; ++j) s += hf[j] * fcw[j];
  for (int j = 0; j < Hh; ++j) s += hbk[j] * fcw[Hh + j];
  out[b] = s;
}

extern "C" void kernel_launch(void* const* d_in, const int* in_sizes, int n_in,
                              void* d_out, int out_size, void* d_ws,
                              size_t ws_size, hipStream_t stream) {
  const float* x    = (const float*)d_in[0];
  const float* Wih0 = (const float*)d_in[1];
  const float* Whh0 = (const float*)d_in[2];
  const float* b0   = (const float*)d_in[3];
  const float* Wih1 = (const float*)d_in[4];
  const float* Whh1 = (const float*)d_in[5];
  const float* b1   = (const float*)d_in[6];
  const float* fcw  = (const float*)d_in[7];
  const float* fcb  = (const float*)d_in[8];
  float* out = (float*)d_out;

  char* ws = (char*)d_ws;
  size_t off = 0;
  auto take = [&](size_t bytes) -> void* {
    off = (off + 255) & ~(size_t)255;
    void* p = ws + off;
    off += bytes;
    return p;
  };
  __bf16* w0p = (__bf16*)take((size_t)2 * NTILES * KT0 * 512 * 2);
  __bf16* w1p = (__bf16*)take((size_t)2 * NTILES * KT1 * 512 * 2);
  __bf16* x0p = (__bf16*)take((size_t)Tt * 16 * XKT0 * 512 * 2);
  __bf16* x1p = (__bf16*)take((size_t)Tt * 16 * XKT1 * 512 * 2);   // ~134 MB
  float*  h1  = (float*)take((size_t)2 * Bb * Hh * 4);

  // --- prep: pack weights + inputs into WMMA operand layouts (bf16) ---
  int tw0 = 2 * NTILES * KT0 * 512;
  pack_w_kernel<<<(tw0 + 255) / 256, 256, 0, stream>>>(Whh0, Wih0, w0p, KT0,
                                                       INx, tw0);
  int tw1 = 2 * NTILES * KT1 * 512;
  pack_w_kernel<<<(tw1 + 255) / 256, 256, 0, stream>>>(Whh1, Wih1, w1p, KT1,
                                                       2 * Hh, tw1);
  int tx0 = Tt * 16 * XKT0 * 512;
  pack_x0_kernel<<<(tx0 + 255) / 256, 256, 0, stream>>>(x, x0p, tx0);

  // --- layer 0: both directions, full T, writes packed h for layer 1 ---
  lstm_layer_kernel<<<dim3(16, 2), 512, 0, stream>>>(
      w0p, x0p, b0, x1p, nullptr, KT0, XKT0, Tt, Tt);

  // --- layer 1: fwd full T, bwd only 1 step (only h[:, -1, :] is needed) ---
  lstm_layer_kernel<<<dim3(16, 2), 512, 0, stream>>>(
      w1p, x1p, b1, nullptr, h1, KT1, XKT1, Tt, 1);

  // --- final FC ---
  fc_kernel<<<1, 256, 0, stream>>>(h1, fcw, fcb, out);
}
